// MultiHeadSelfAttention_438086664265
// MI455X (gfx1250) — matrix operand order repaired and checked by reading, not compiled, not run on hardware
//
#include <hip/hip_runtime.h>
#include <hip/hip_bf16.h>

typedef __bf16 bf16_t;
typedef __attribute__((ext_vector_type(16))) __bf16 v16bf;
typedef __attribute__((ext_vector_type(8)))  __bf16 v8bf;
typedef __attribute__((ext_vector_type(8)))  float  v8f;

#define B_   4
#define S_   2048
#define D_   1024
#define H_   16
#define DK_  64
#define M_   (B_ * S_)   // 8192 rows
#define E3_  (3 * D_)    // 3072

#define APAD 40          // padded LDS row stride (elements) for 32-wide tiles
#define AS3  __attribute__((address_space(3)))

// ---------------------------------------------------------------------------
// Async Global->LDS copy (CDNA5, tracked by ASYNCcnt). 16 bytes per lane.
// vdst operand carries the per-lane LDS byte offset, vaddr the 64-bit global
// address (CDNA5 ISA / CDNA5 ISA opcode 98).
// ---------------------------------------------------------------------------
__device__ __forceinline__ void async_g2l_b128(bf16_t* lds_dst, const bf16_t* g) {
    unsigned loff = (unsigned)(unsigned long long)(AS3 bf16_t*)lds_dst;
    asm volatile("global_load_async_to_lds_b128 %0, %1, off"
                 :: "v"(loff), "v"(g) : "memory");
}
__device__ __forceinline__ void wait_async0() {
    asm volatile("s_wait_asynccnt 0" ::: "memory");
}

// ---------------------------------------------------------------------------
// Fragment loaders following the CDNA5 ISA VGPR layouts (CDNA5 ISA)
// A-matrix 16x32 bf16: lane m = lane&15, half = lane>>4.
//   elements 0..7  -> K = 8*half + e          (contiguous 16B)
//   elements 8..15 -> K = 16 + 8*half + (e-8) (contiguous 16B)
// Works for both global and LDS (generic) pointers.
// ---------------------------------------------------------------------------
__device__ __forceinline__ v16bf load_a_frag(const bf16_t* A, int lda, int m0,
                                             int k0, int lane) {
    const int m  = m0 + (lane & 15);
    const int hf = lane >> 4;
    const bf16_t* p = A + (size_t)m * lda + k0 + hf * 8;
    v8bf lo = *(const v8bf*)(p);
    v8bf hi = *(const v8bf*)(p + 16);
    v16bf r;
#pragma unroll
    for (int i = 0; i < 8; ++i) { r[i] = lo[i]; r[8 + i] = hi[i]; }
    return r;
}

// B-matrix 32x16 bf16 (Bn given as N rows x K contiguous):
//   lane n = lane&15, half = lane>>4; elements 0..7 -> K = 8*half + e,
//   elements 8..15 -> K = 16 + 8*half + (e-8) (same order as A)
__device__ __forceinline__ v16bf load_b_frag(const bf16_t* Bn, int ldb, int n0,
                                             int k0, int lane) {
    const int n  = n0 + (lane & 15);
    const int hf = lane >> 4;
    const bf16_t* p = Bn + (size_t)n * ldb + k0 + hf * 8;
    v8bf lo = *(const v8bf*)(p);
    v8bf hi = *(const v8bf*)(p + 16);
    v16bf r;
#pragma unroll
    for (int i = 0; i < 8; ++i) { r[i] = lo[i]; r[8 + i] = hi[i]; }
    return r;
}

__device__ __forceinline__ v8f wmma_bf16(v16bf a, v16bf b, v8f c) {
    return __builtin_amdgcn_wmma_f32_16x16x32_bf16(
        /*neg_a=*/false, a, /*neg_b=*/false, b,
        /*c_mod=*/(short)0, c, /*reuse_a=*/false, /*reuse_b=*/false);
}

// ---------------------------------------------------------------------------
// f32 -> bf16 conversion
// ---------------------------------------------------------------------------
__global__ void cvt_f32_bf16(const float* __restrict__ in,
                             bf16_t* __restrict__ out, int n) {
    int i = blockIdx.x * blockDim.x + threadIdx.x;
    if (i < n) out[i] = (bf16_t)in[i];
}

// ---------------------------------------------------------------------------
// Shared GEMM core with async double-buffered LDS staging.
// Block: 256 threads / 8 waves. Block tile 64(M) x 256(N), K-step 32.
// Per K-step: stage A 64x32 (4KB) + B 256x32 (16KB) via async b128,
// overlap with WMMA on the other buffer. Wave tile 32x64 (8 accum tiles).
// ---------------------------------------------------------------------------
struct GemmFrags { v8f acc[2][4]; };

__device__ __forceinline__ void gemm_core(const bf16_t* __restrict__ Aglob,
                                          const bf16_t* __restrict__ Bglob,
                                          int lda, int ldb,
                                          int blockM0, int blockN0,
                                          bf16_t* AshBase, bf16_t* BshBase,
                                          GemmFrags& f) {
    const int tid  = threadIdx.x;
    const int lane = tid & 31;
    const int w    = tid >> 5;
    const int m0w  = (w & 1) * 32;   // wave offset in block M tile
    const int n0w  = (w >> 1) * 64;  // wave offset in block N tile

#pragma unroll
    for (int i = 0; i < 2; ++i)
#pragma unroll
        for (int j = 0; j < 4; ++j) f.acc[i][j] = v8f{};

    const int ar = tid >> 2, ac = tid & 3;   // A: 64 rows x 4 chunks

    // prologue: stage K-step 0 into buffer 0
    async_g2l_b128(AshBase + ar * APAD + ac * 8,
                   Aglob + (size_t)(blockM0 + ar) * lda + ac * 8);
#pragma unroll
    for (int t = 0; t < 4; ++t) {
        int ci = t * 256 + tid, br = ci >> 2, bc = ci & 3;
        async_g2l_b128(BshBase + br * APAD + bc * 8,
                       Bglob + (size_t)(blockN0 + br) * ldb + bc * 8);
    }

    for (int k0 = 0; k0 < D_; k0 += 32) {
        const int cur = (k0 >> 5) & 1;
        bf16_t* Acur = AshBase + cur * (64 * APAD);
        bf16_t* Bcur = BshBase + cur * (256 * APAD);

        wait_async0();       // our slice of 'cur' has landed
        __syncthreads();     // everyone's slice has landed

        if (k0 + 32 < D_) {  // stage next K-step into the other buffer
            bf16_t* Anxt = AshBase + (1 - cur) * (64 * APAD);
            bf16_t* Bnxt = BshBase + (1 - cur) * (256 * APAD);
            async_g2l_b128(Anxt + ar * APAD + ac * 8,
                           Aglob + (size_t)(blockM0 + ar) * lda + k0 + 32 + ac * 8);
#pragma unroll
            for (int t = 0; t < 4; ++t) {
                int ci = t * 256 + tid, br = ci >> 2, bc = ci & 3;
                async_g2l_b128(Bnxt + br * APAD + bc * 8,
                               Bglob + (size_t)(blockN0 + br) * ldb + k0 + 32 + bc * 8);
            }
        }

        v16bf a0 = load_a_frag(Acur, APAD, m0w, 0, lane);
        v16bf a1 = load_a_frag(Acur, APAD, m0w + 16, 0, lane);
#pragma unroll
        for (int j = 0; j < 4; ++j) {
            v16bf b = load_b_frag(Bcur, APAD, n0w + 16 * j, 0, lane);
            f.acc[0][j] = wmma_bf16(a0, b, f.acc[0][j]);
            f.acc[1][j] = wmma_bf16(a1, b, f.acc[1][j]);
        }
        __syncthreads();     // done reading 'cur' before it is overwritten
    }
}

// ---------------------------------------------------------------------------
// QKV projection: qkv[m,e] = sum_d x[m,d] * Wqkv[e,d]
// M=8192, N=3072, K=1024. Scatter into Q[B,H,S,DK], K[B,H,S,DK], Vt[B,H,DK,S].
// ---------------------------------------------------------------------------
__global__ void __launch_bounds__(256) qkv_gemm(const bf16_t* __restrict__ X,
                                                const bf16_t* __restrict__ W,
                                                bf16_t* __restrict__ Q,
                                                bf16_t* __restrict__ K,
                                                bf16_t* __restrict__ Vt) {
    __shared__ __align__(16) bf16_t Ash[2][64 * APAD];
    __shared__ __align__(16) bf16_t Bsh[2][256 * APAD];

    const int lane = threadIdx.x & 31;
    const int w    = threadIdx.x >> 5;
    const int m0   = blockIdx.x * 64 + (w & 1) * 32;
    const int n0   = blockIdx.y * 256 + (w >> 1) * 64;

    GemmFrags f;
    gemm_core(X, W, D_, D_, blockIdx.x * 64, blockIdx.y * 256,
              &Ash[0][0], &Bsh[0][0], f);

    const int hf = lane >> 4;
    const int nl = lane & 15;
#pragma unroll
    for (int i = 0; i < 2; ++i)
#pragma unroll
        for (int j = 0; j < 4; ++j)
#pragma unroll
            for (int r = 0; r < 8; ++r) {
                int m  = m0 + 16 * i + hf * 8 + r;   // global row = b*S + s
                int n  = n0 + 16 * j + nl;           // e = c*D + h*DK + dk
                int c  = n >> 10;                    // /D_
                int e  = n & (D_ - 1);
                int h  = e >> 6;                     // /DK_
                int dk = e & (DK_ - 1);
                int bb = m >> 11;                    // /S_
                int s  = m & (S_ - 1);
                bf16_t v  = (bf16_t)f.acc[i][j][r];
                size_t bh = (size_t)bb * H_ + h;
                if (c == 0)      Q [(bh * S_ + s) * DK_ + dk] = v;
                else if (c == 1) K [(bh * S_ + s) * DK_ + dk] = v;
                else             Vt[(bh * DK_ + dk) * S_ + s] = v;  // transposed
            }
}

// ---------------------------------------------------------------------------
// Flash attention (causal). One wave handles one 16-row Q tile for one (b,h).
// kv processed in 32-key blocks: 2x score WMMA (N-tiles), online softmax,
// P staged C-layout -> A-layout via per-wave LDS, then 4x P.V WMMA (K=32).
// ---------------------------------------------------------------------------
__global__ void __launch_bounds__(256) flash_attn(const bf16_t* __restrict__ Q,
                                                  const bf16_t* __restrict__ K,
                                                  const bf16_t* __restrict__ Vt,
                                                  bf16_t* __restrict__ Hd) {
    __shared__ __align__(64) bf16_t pbuf[8][16 * 32];  // per-wave P tile

    const int lane = threadIdx.x & 31;
    const int w    = threadIdx.x >> 5;
    const int bh   = blockIdx.x;            // 0 .. B*H-1
    const int qt   = blockIdx.y * 8 + w;    // 0 .. 127
    const int q0   = qt * 16;

    const bf16_t* Qh  = Q  + (size_t)bh * S_ * DK_;
    const bf16_t* Kh  = K  + (size_t)bh * S_ * DK_;
    const bf16_t* Vth = Vt + (size_t)bh * S_ * DK_;

    // Q fragments for both DK k-steps, kept in registers for the whole row.
    v16bf aq0 = load_a_frag(Qh, DK_, q0, 0, lane);
    v16bf aq1 = load_a_frag(Qh, DK_, q0, 32, lane);

    v8f o[4] = {};
    float mrow[8], lrow[8];
#pragma unroll
    for (int r = 0; r < 8; ++r) { mrow[r] = -1e30f; lrow[r] = 0.0f; }

    const int hf = lane >> 4;
    const int nl = lane & 15;
    bf16_t* pb = &pbuf[w][0];

    for (int kv0 = 0; kv0 < q0 + 16; kv0 += 32) {
        if (kv0 + 32 < q0 + 16) {  // near-scope prefetch of next K block
            __builtin_prefetch(Kh + (size_t)(kv0 + 32 + nl) * DK_, 0, 3);
        }
        // scores: two 16x16 tiles, K-dim = DK = 64 (2 WMMA steps each)
        v8f sc0 = {};
        v8f sc1 = {};
        sc0 = wmma_bf16(aq0, load_b_frag(Kh, DK_, kv0,      0,  lane), sc0);
        sc0 = wmma_bf16(aq1, load_b_frag(Kh, DK_, kv0,      32, lane), sc0);
        sc1 = wmma_bf16(aq0, load_b_frag(Kh, DK_, kv0 + 16, 0,  lane), sc1);
        sc1 = wmma_bf16(aq1, load_b_frag(Kh, DK_, kv0 + 16, 32, lane), sc1);

        // online softmax per row (row r of this lane-half), causal mask
#pragma unroll
        for (int r = 0; r < 8; ++r) {
            int   mg = q0 + hf * 8 + r;
            float s0 = sc0[r] * 0.125f;   // 1/sqrt(64)
            float s1 = sc1[r] * 0.125f;
            if (kv0 + nl > mg)      s0 = -1e30f;
            if (kv0 + 16 + nl > mg) s1 = -1e30f;
            float mx = fmaxf(s0, s1);
#pragma unroll
            for (int d = 1; d < 16; d <<= 1)
                mx = fmaxf(mx, __shfl_xor(mx, d, 16));
            float mnew = fmaxf(mrow[r], mx);
            float p0 = __expf(s0 - mnew);
            float p1 = __expf(s1 - mnew);
            float ps = p0 + p1;
#pragma unroll
            for (int d = 1; d < 16; d <<= 1)
                ps += __shfl_xor(ps, d, 16);
            float alpha = __expf(mrow[r] - mnew);
            lrow[r] = lrow[r] * alpha + ps;
            mrow[r] = mnew;
#pragma unroll
            for (int j = 0; j < 4; ++j) o[j][r] *= alpha;
            int m = hf * 8 + r;
            pb[m * 32 + nl]      = (bf16_t)p0;   // C-layout -> [m][key] in LDS
            pb[m * 32 + 16 + nl] = (bf16_t)p1;
        }
        asm volatile("s_wait_dscnt 0" ::: "memory");  // P stores visible

        // P (16x32, A-layout from LDS) x V (32 keys x 64 cols, from Vt rows)
        v16bf pa = load_a_frag(pb, 32, 0, 0, lane);
#pragma unroll
        for (int j = 0; j < 4; ++j) {
            v16bf bv = load_b_frag(Vth, S_, 16 * j, kv0, lane);
            o[j] = wmma_bf16(pa, bv, o[j]);
        }
    }

    // normalize and store merged heads [B, S, D] as bf16
    const int bb = bh / H_;
    const int h  = bh % H_;
#pragma unroll
    for (int j = 0; j < 4; ++j)
#pragma unroll
        for (int r = 0; r < 8; ++r) {
            int s   = q0 + hf * 8 + r;
            int col = h * DK_ + 16 * j + nl;
            Hd[((size_t)bb * S_ + s) * D_ + col] = (bf16_t)(o[j][r] / lrow[r]);
        }
}

// ---------------------------------------------------------------------------
// Output projection: out[m,d] = sum_e heads[m,e] * Wo[d,e]. f32 output.
// ---------------------------------------------------------------------------
__global__ void __launch_bounds__(256) out_gemm(const bf16_t* __restrict__ Hd,
                                                const bf16_t* __restrict__ Wo,
                                                float* __restrict__ Out) {
    __shared__ __align__(16) bf16_t Ash[2][64 * APAD];
    __shared__ __align__(16) bf16_t Bsh[2][256 * APAD];

    const int lane = threadIdx.x & 31;
    const int w    = threadIdx.x >> 5;
    const int m0   = blockIdx.x * 64 + (w & 1) * 32;
    const int n0   = blockIdx.y * 256 + (w >> 1) * 64;

    GemmFrags f;
    gemm_core(Hd, Wo, D_, D_, blockIdx.x * 64, blockIdx.y * 256,
              &Ash[0][0], &Bsh[0][0], f);

    const int hf = lane >> 4;
    const int nl = lane & 15;
#pragma unroll
    for (int i = 0; i < 2; ++i)
#pragma unroll
        for (int j = 0; j < 4; ++j)
#pragma unroll
            for (int r = 0; r < 8; ++r) {
                int m = m0 + 16 * i + hf * 8 + r;
                int n = n0 + 16 * j + nl;
                Out[(size_t)m * D_ + n] = f.acc[i][j][r];
            }
}

// ---------------------------------------------------------------------------
// Workspace layout (bytes):
//   xb    @ 0         : 8192*1024 bf16  = 16 MB
//   wqkvb @ 16777216  : 3072*1024 bf16  =  6 MB
//   wob   @ 23068672  : 1024*1024 bf16  =  2 MB
//   Qb    @ 25165824  : 16 MB   [B,H,S,DK]
//   Kb    @ 41943040  : 16 MB   [B,H,S,DK]
//   Vtb   @ 58720256  : 16 MB   [B,H,DK,S] (transposed)
//   Hd    @ 75497472  : 16 MB   [B,S,D]
//   total ~88 MB  (fits in 192 MB L2 alongside in/out)
// ---------------------------------------------------------------------------
extern "C" void kernel_launch(void* const* d_in, const int* in_sizes, int n_in,
                              void* d_out, int out_size, void* d_ws, size_t ws_size,
                              hipStream_t stream) {
    (void)in_sizes; (void)n_in; (void)out_size; (void)ws_size;

    const float* x    = (const float*)d_in[0];
    const float* Wqkv = (const float*)d_in[1];
    const float* Wo   = (const float*)d_in[2];
    float*       out  = (float*)d_out;

    char* ws = (char*)d_ws;
    bf16_t* xb    = (bf16_t*)(ws);
    bf16_t* wqkvb = (bf16_t*)(ws + 16777216);
    bf16_t* wob   = (bf16_t*)(ws + 23068672);
    bf16_t* Qb    = (bf16_t*)(ws + 25165824);
    bf16_t* Kb    = (bf16_t*)(ws + 41943040);
    bf16_t* Vtb   = (bf16_t*)(ws + 58720256);
    bf16_t* Hd    = (bf16_t*)(ws + 75497472);

    const int nx = M_ * D_;     // 8388608
    const int nw = E3_ * D_;    // 3145728
    const int no = D_ * D_;     // 1048576

    cvt_f32_bf16<<<(nx + 255) / 256, 256, 0, stream>>>(x, xb, nx);
    cvt_f32_bf16<<<(nw + 255) / 256, 256, 0, stream>>>(Wqkv, wqkvb, nw);
    cvt_f32_bf16<<<(no + 255) / 256, 256, 0, stream>>>(Wo, wob, no);

    qkv_gemm<<<dim3(M_ / 64, E3_ / 256), 256, 0, stream>>>(xb, wqkvb, Qb, Kb, Vtb);
    flash_attn<<<dim3(B_ * H_, S_ / 128), 256, 0, stream>>>(Qb, Kb, Vtb, Hd);
    out_gemm<<<dim3(M_ / 64, D_ / 256), 256, 0, stream>>>(Hd, wob, out);
}
